// SelfAttention_54065048322683
// MI455X (gfx1250) — compile-verified, matrix operand order checked by reading, not run on hardware
//
#include <hip/hip_runtime.h>

typedef _Float16 v16h __attribute__((ext_vector_type(16)));
typedef _Float16 v8h  __attribute__((ext_vector_type(8)));
typedef float    v8f  __attribute__((ext_vector_type(8)));
typedef int      v4i  __attribute__((ext_vector_type(4)));

#define B_  4
#define S_  2048
#define D_  1024
#define H_  16
#define HD_ 64

// CDNA5 async global->LDS path (probe via __has_builtin, sync fallback otherwise)
#if defined(__has_builtin)
#if __has_builtin(__builtin_amdgcn_global_load_async_to_lds_b128) && \
    __has_builtin(__builtin_amdgcn_s_wait_asynccnt)
#define ASYNC_LDS 1
#endif
#endif

// builtin signature (from hipcc diagnostic): (v4i addrspace(1)*, v4i addrspace(3)*, imm, imm)
typedef __attribute__((address_space(1))) v4i gv4i_t;
typedef __attribute__((address_space(3))) v4i lv4i_t;

static __device__ __forceinline__ void async_copy16(_Float16* ldst, const _Float16* gsrc) {
#if defined(ASYNC_LDS)
    __builtin_amdgcn_global_load_async_to_lds_b128((gv4i_t*)gsrc, (lv4i_t*)ldst, 0, 0);
#else
    *reinterpret_cast<v8h*>(ldst) = *reinterpret_cast<const v8h*>(gsrc);
#endif
}

// WMMA f16 operand layout (16x16x32, wave32):
//   A (16xK): lane m = row; element i -> k = i + 8*((i/8) + (lane/16))
//   B (Kx16): lane n = col; same k packing.
// Each lane's operand = two contiguous 8-half chunks at +8*lh and +16+8*lh.
static __device__ __forceinline__ v16h load_ab_f16(const _Float16* __restrict__ row, int lh) {
    v8h c0 = *reinterpret_cast<const v8h*>(row + 8 * lh);
    v8h c1 = *reinterpret_cast<const v8h*>(row + 16 + 8 * lh);
    return __builtin_shufflevector(c0, c1, 0,1,2,3,4,5,6,7,8,9,10,11,12,13,14,15);
}

static __device__ __forceinline__ v8f wmma16(v16h a, v16h b, v8f c) {
    return __builtin_amdgcn_wmma_f32_16x16x32_f16(false, a, false, b, (short)0, c,
                                                  false, false);
}

// ---------------- one-shot fp32 -> f16 cast (8 elements / thread) -------------
__global__ __launch_bounds__(256) void cast8_kernel(const float* __restrict__ src,
                                                    _Float16* __restrict__ dst, int n8) {
    const int i = blockIdx.x * 256 + threadIdx.x;
    if (i < n8) {
        v8f v = *reinterpret_cast<const v8f*>(src + (size_t)i * 8);
        *reinterpret_cast<v8h*>(dst + (size_t)i * 8) = __builtin_convertvector(v, v8h);
    }
}

// ---------------- Q/K projection: D[j, s] = sum_d W[j,d] * X[s,d] + bias[j] ----
// A = W rows (M = feature j), B = X^T (N = token s). Stores f16 [B,H,S,HD].
__global__ __launch_bounds__(128) void proj_qk_kernel(
    const _Float16* __restrict__ X, const _Float16* __restrict__ W,
    const float* __restrict__ bias, _Float16* __restrict__ out) {
    const int lane = threadIdx.x & 31;
    const int wv   = threadIdx.x >> 5;
    const int lh   = lane >> 4;
    const int ln   = lane & 15;
    const int jt     = blockIdx.y;            // 0..63  (16 features each)
    const int sblock = blockIdx.x * 4 + wv;   // 0..127 (64 tokens each)

    const _Float16* wrow = W + (size_t)(jt * 16 + ln) * D_;
    const _Float16* xrow[4];
#pragma unroll
    for (int sub = 0; sub < 4; ++sub)
        xrow[sub] = X + (size_t)(sblock * 64 + sub * 16 + ln) * D_;

    v8f acc[4] = {};
    for (int kk = 0; kk < D_; kk += 32) {
        v16h a = load_ab_f16(wrow + kk, lh);
#pragma unroll
        for (int sub = 0; sub < 4; ++sub) {
            v16h bb = load_ab_f16(xrow[sub] + kk, lh);
            acc[sub] = wmma16(a, bb, acc[sub]);
        }
    }

    const int j0  = jt * 16 + 8 * lh;           // 8 contiguous features per lane
    v8f bvec = *reinterpret_cast<const v8f*>(bias + j0);
    const int h   = j0 >> 6;
    const int hd0 = j0 & 63;
#pragma unroll
    for (int sub = 0; sub < 4; ++sub) {
        v8f r = acc[sub] + bvec;
        v8h o = __builtin_convertvector(r, v8h);
        const int n = sblock * 64 + sub * 16 + ln;   // global token row
        const int b = n >> 11;
        const int s = n & (S_ - 1);
        _Float16* p = out + (((size_t)(b * H_ + h) * S_ + s) * HD_ + hd0);
        *reinterpret_cast<v8h*>(p) = o;
    }
}

// ---------------- V projection: D[s, hd] = sum_d X[s,d] * Wv[h*64+hd, d] + bv ----
// A = X rows (M = token s), B = Wv^T (N = hd). Stores transposed f16 Vt[B,H,HD,S].
__global__ __launch_bounds__(128) void proj_v_kernel(
    const _Float16* __restrict__ X, const _Float16* __restrict__ W,
    const float* __restrict__ bias, _Float16* __restrict__ outT) {
    const int lane = threadIdx.x & 31;
    const int wv   = threadIdx.x >> 5;
    const int lh   = lane >> 4;
    const int ln   = lane & 15;
    const int st = blockIdx.x * 4 + wv;   // 0..511 token tiles (16 each)
    const int h  = blockIdx.y;            // head

    const _Float16* xrow = X + (size_t)(st * 16 + ln) * D_;
    const _Float16* wrow[4];
    float bsc[4];
#pragma unroll
    for (int sub = 0; sub < 4; ++sub) {
        const int j = h * HD_ + sub * 16 + ln;
        wrow[sub] = W + (size_t)j * D_;
        bsc[sub]  = bias[j];
    }

    v8f acc[4] = {};
    for (int kk = 0; kk < D_; kk += 32) {
        v16h a = load_ab_f16(xrow + kk, lh);
#pragma unroll
        for (int sub = 0; sub < 4; ++sub) {
            v16h bb = load_ab_f16(wrow[sub] + kk, lh);
            acc[sub] = wmma16(a, bb, acc[sub]);
        }
    }

    const int n0 = st * 16 + 8 * lh;      // 8 contiguous tokens per lane
    const int b  = (st * 16) >> 11;
    const int s0 = n0 & (S_ - 1);
#pragma unroll
    for (int sub = 0; sub < 4; ++sub) {
        v8f r = acc[sub] + bsc[sub];
        v8h o = __builtin_convertvector(r, v8h);
        const int hd = sub * 16 + ln;
        _Float16* p = outT + (((size_t)(b * H_ + h) * HD_ + hd) * S_ + s0);
        *reinterpret_cast<v8h*>(p) = o;
    }
}

// ---------------- Flash attention, LDS-staged K/V, double buffered -------------
// 4 waves/block share one (b,h): K chunk (32x64) + Vt chunk (64x32) staged via
// async global->LDS. St[k,q] = K.Q^T/8 (+mask); online softmax per lane; the St
// accumulator layout feeds P^T directly as B operand of ctx^T += V^T * P^T.
#define KCH  32
#define KROW 72   // 64 + 8 halves pad (16B-aligned rows, de-conflicted banks)
#define VROW 40   // 32 + 8 halves pad

__global__ __launch_bounds__(128) void attn_kernel(
    const _Float16* __restrict__ Qh, const _Float16* __restrict__ Kh,
    const _Float16* __restrict__ Vt, const float* __restrict__ mask,
    float* __restrict__ out) {
    __shared__ _Float16 sK[2][KCH * KROW];
    __shared__ _Float16 sV[2][HD_ * VROW];

    const int t    = threadIdx.x;
    const int lane = t & 31;
    const int wv   = t >> 5;
    const int lh   = lane >> 4;
    const int ln   = lane & 15;
    const int qt = blockIdx.x * 4 + wv;   // 0..127 query tiles (16 each)
    const int h  = blockIdx.y;
    const int b  = blockIdx.z;

    const size_t bh = (size_t)(b * H_ + h);
    const _Float16* qbase = Qh + bh * S_ * HD_;
    const _Float16* kbase = Kh + bh * S_ * HD_;
    const _Float16* vbase = Vt + bh * HD_ * S_;
    const float*    mrow  = mask + (size_t)b * S_;

    // cooperative stage of one 32-key chunk: 4 x b128 async ops per thread
    auto stage = [&](int buf, int kb) {
#pragma unroll
        for (int i = 0; i < 2; ++i) {            // K: 32 rows x 8 chunks of 8 halves
            const int e = t + i * 128, row = e >> 3, ch = e & 7;
            async_copy16(&sK[buf][row * KROW + ch * 8],
                         kbase + (size_t)(kb + row) * HD_ + ch * 8);
        }
#pragma unroll
        for (int i = 0; i < 2; ++i) {            // Vt: 64 rows x 4 chunks of 8 halves
            const int e = t + i * 128, row = e >> 2, ch = e & 3;
            async_copy16(&sV[buf][row * VROW + ch * 8],
                         vbase + (size_t)row * S_ + kb + ch * 8);
        }
    };

    const int q = qt * 16 + ln;           // this lane's query column
    const v16h bq0 = load_ab_f16(qbase + (size_t)q * HD_, lh);       // d 0..31
    const v16h bq1 = load_ab_f16(qbase + (size_t)q * HD_ + 32, lh);  // d 32..63

    float m_run = -__builtin_inff();
    float l_run = 0.f;
    v8f ctx[4] = {};

    stage(0, 0);
    for (int it = 0; it < S_ / KCH; ++it) {
        const int  kb   = it * KCH;
        const int  cur  = it & 1;
        const bool more = (it + 1) < (S_ / KCH);
        if (more) stage(cur ^ 1, kb + KCH);
#if defined(ASYNC_LDS)
        if (more) __builtin_amdgcn_s_wait_asynccnt(4);  // in-order: prev chunk landed
        else      __builtin_amdgcn_s_wait_asynccnt(0);
#endif
        __syncthreads();                                // all waves' stage visible

        // ---- scores: two 16-key tiles from LDS, contraction over HD=64
        const _Float16* k0 = &sK[cur][ln * KROW];
        const _Float16* k1 = &sK[cur][(16 + ln) * KROW];
        v8f st0 = {}, st1 = {};
        st0 = wmma16(load_ab_f16(k0, lh),      bq0, st0);
        st0 = wmma16(load_ab_f16(k0 + 32, lh), bq1, st0);
        st1 = wmma16(load_ab_f16(k1, lh),      bq0, st1);
        st1 = wmma16(load_ab_f16(k1 + 32, lh), bq1, st1);

        // ---- scale + custom mask (key-indexed -> contiguous per lane)
        v8f m0 = *reinterpret_cast<const v8f*>(mrow + kb + 8 * lh);
        v8f m1 = *reinterpret_cast<const v8f*>(mrow + kb + 16 + 8 * lh);
        float s0[8], s1[8];
        float cmax = -__builtin_inff();
#pragma unroll
        for (int r = 0; r < 8; ++r) {
            float a = st0[r] * 0.125f;
            a = (m0[r] >= 0.f ? a : 0.f) + m0[r];
            s0[r] = a; cmax = fmaxf(cmax, a);
            float c = st1[r] * 0.125f;
            c = (m1[r] >= 0.f ? c : 0.f) + m1[r];
            s1[r] = c; cmax = fmaxf(cmax, c);
        }
        cmax = fmaxf(cmax, __shfl_xor(cmax, 16));   // combine key-half lanes
        const float m_new = fmaxf(m_run, cmax);
        const float alpha = __expf(m_run - m_new);

        float lsum = 0.f;
        v16h pb;   // P^T as B operand: st layout == B k-packing, no shuffles
#pragma unroll
        for (int r = 0; r < 8; ++r) {
            float p0 = __expf(s0[r] - m_new);
            float p1 = __expf(s1[r] - m_new);
            lsum += p0 + p1;
            pb[r]     = (_Float16)p0;
            pb[r + 8] = (_Float16)p1;
        }
        lsum += __shfl_xor(lsum, 16);
        l_run = l_run * alpha + lsum;
        m_run = m_new;

        // ---- ctx^T += V^T (16 d-rows x 32 k) * P^T (32 k x 16 q), V from LDS
#pragma unroll
        for (int dt = 0; dt < 4; ++dt) {
            v8f c = ctx[dt] * alpha;
            ctx[dt] = wmma16(load_ab_f16(&sV[cur][(dt * 16 + ln) * VROW], lh), pb, c);
        }
        __syncthreads();            // all reads of buffer `cur` done before refill
    }

    const float inv = 1.f / l_run;
    float* obase = out + ((size_t)(b * S_ + q) * D_ + h * HD_);
#pragma unroll
    for (int dt = 0; dt < 4; ++dt) {
        v8f r = ctx[dt] * inv;
        *reinterpret_cast<v8f*>(obase + dt * 16 + 8 * lh) = r;   // contiguous in d
    }
}

extern "C" void kernel_launch(void* const* d_in, const int* in_sizes, int n_in,
                              void* d_out, int out_size, void* d_ws, size_t ws_size,
                              hipStream_t stream) {
    (void)in_sizes; (void)n_in; (void)out_size; (void)ws_size;
    const float* X    = (const float*)d_in[0];
    const float* mask = (const float*)d_in[1];
    const float* Wq   = (const float*)d_in[2];
    const float* bq   = (const float*)d_in[3];
    const float* Wk   = (const float*)d_in[4];
    const float* bk   = (const float*)d_in[5];
    const float* Wv   = (const float*)d_in[6];
    const float* bv   = (const float*)d_in[7];
    float* out = (float*)d_out;

    const size_t nTok = (size_t)B_ * S_ * D_;   // 8M elements
    const size_t nW   = (size_t)D_ * D_;        // 1M elements
    _Float16* Qh  = (_Float16*)d_ws;
    _Float16* Kh  = Qh + nTok;
    _Float16* Vt  = Kh + nTok;
    _Float16* Xh  = Vt + nTok;
    _Float16* Wqh = Xh + nTok;
    _Float16* Wkh = Wqh + nW;
    _Float16* Wvh = Wkh + nW;

    // one-shot f16 casts (removes redundant per-tile cvt work from GEMM loops)
    cast8_kernel<<<dim3((int)(nTok / 8 / 256)), 256, 0, stream>>>(X,  Xh,  (int)(nTok / 8));
    cast8_kernel<<<dim3((int)(nW   / 8 / 256)), 256, 0, stream>>>(Wq, Wqh, (int)(nW / 8));
    cast8_kernel<<<dim3((int)(nW   / 8 / 256)), 256, 0, stream>>>(Wk, Wkh, (int)(nW / 8));
    cast8_kernel<<<dim3((int)(nW   / 8 / 256)), 256, 0, stream>>>(Wv, Wvh, (int)(nW / 8));

    dim3 blk(128);  // 4 waves of 32
    proj_qk_kernel<<<dim3(32, 64),    blk, 0, stream>>>(Xh, Wqh, bq, Qh);
    proj_qk_kernel<<<dim3(32, 64),    blk, 0, stream>>>(Xh, Wkh, bk, Kh);
    proj_v_kernel <<<dim3(128, 16),   blk, 0, stream>>>(Xh, Wvh, bv, Vt);
    attn_kernel   <<<dim3(32, 16, 4), blk, 0, stream>>>(Qh, Kh, Vt, mask, out);
}
